// DiscriminativeLoss_89215060673003
// MI455X (gfx1250) — matrix operand order repaired and checked by reading, not compiled, not run on hardware
//
#include <hip/hip_runtime.h>
#include <hip/hip_bf16.h>
#include <math.h>

// Problem constants (fixed by the reference).
constexpr int Bc = 8;
constexpr int Nc = 4096;
constexpr int Ec = 64;
constexpr int Kc = 32;
constexpr float DELTA_V = 0.5f;
constexpr float MARGIN_D = 3.0f;   // 2 * DELTA_D
constexpr float GAMMA_W = 0.001f;

typedef __attribute__((ext_vector_type(16))) _Float16 v16h;
typedef __attribute__((ext_vector_type(8)))  float    v8f;

// ---------------------------------------------------------------------------
// Kernel A: per-batch centroid sums via WMMA, then counts and centroids.
// grid = B blocks, 256 threads (8 waves). Wave w owns the 16x16 output tile
// (ko = w/4 selects K rows 0-15/16-31, eo = w%4 selects E cols eo*16..+15).
// D = A(16x32 f16: masks^T tile) x B(32x16 f16: embedded tile) accumulated
// over N in chunks of 32 -> 128 v_wmma_f32_16x16x32_f16 per wave.
// ---------------------------------------------------------------------------
__global__ void __launch_bounds__(256)
centroid_wmma_kernel(const float* __restrict__ emb,
                     const float* __restrict__ masks,
                     float* __restrict__ cent,      // [B,K,E]
                     float* __restrict__ msum_tot,  // [B]
                     float* __restrict__ var_acc)   // [B] (zeroed here)
{
    const int b    = blockIdx.x;
    const int tid  = threadIdx.x;
    const int lane = tid & 31;
    const int wave = tid >> 5;      // 0..7
    const int ko   = wave >> 2;     // 0..1
    const int eo   = wave & 3;      // 0..3
    const int m    = lane & 15;     // A row (cluster within ko block); also B/D col
    const int h    = lane >> 4;     // lane half selects K sub-range

    __shared__ float S[Kc][Ec + 1];     // centroid sums (padded)
    __shared__ float red[256];
    __shared__ float msumS[Kc];

    const float* __restrict__ embB = emb   + (size_t)b * Nc * Ec;
    const float* __restrict__ mskB = masks + (size_t)b * Nc * Kc;

    v8f acc = {0.f, 0.f, 0.f, 0.f, 0.f, 0.f, 0.f, 0.f};

    for (int n0 = 0; n0 < Nc; n0 += 32) {
        v16h afrag, bfrag;
#pragma unroll
        for (int i = 0; i < 16; ++i) {
            // A fragment (16x32 f16 layout): contraction index within chunk.
            const int kkA = i + 8 * h + ((i >= 8) ? 8 : 0);
            afrag[i] = (_Float16)mskB[(size_t)(n0 + kkA) * Kc + ko * 16 + m];
            // B fragment (32x16 f16 layout): same K order as A (8h+i, then 16+8h+(i-8)).
            const int kkB = i + 8 * h + ((i >= 8) ? 8 : 0);
            bfrag[i] = (_Float16)embB[(size_t)(n0 + kkB) * Ec + eo * 16 + m];
        }
        acc = __builtin_amdgcn_wmma_f32_16x16x32_f16(
            /*neg_a=*/false, afrag, /*neg_b=*/false, bfrag,
            /*c_mod=*/(short)0, acc, /*reuse_a=*/false, /*reuse_b=*/false);
    }

    // D tile -> LDS: lane half h gives rows r (h==0) or r+8 (h==1).
#pragma unroll
    for (int r = 0; r < 8; ++r)
        S[ko * 16 + r + 8 * h][eo * 16 + m] = acc[r];

    // Per-cluster counts: thread t sums masks over an N-chunk for k = t%32.
    {
        const int k     = tid & 31;
        const int chunk = tid >> 5;               // 0..7 -> 512 points each
        float s = 0.f;
        const int nBeg = chunk * (Nc / 8), nEnd = nBeg + (Nc / 8);
        for (int n = nBeg; n < nEnd; ++n) s += mskB[(size_t)n * Kc + k];
        red[tid] = s;
    }
    __syncthreads();
    if (tid < Kc) {
        float s = 0.f;
#pragma unroll
        for (int c = 0; c < 8; ++c) s += red[c * 32 + tid];
        msumS[tid] = s;
    }
    __syncthreads();

    if (tid == 0) {
        float tot = 0.f;
#pragma unroll
        for (int k = 0; k < Kc; ++k) tot += msumS[k];
        msum_tot[b] = tot;       // == N for one-hot masks
        var_acc[b]  = 0.f;       // init accumulator for kernel B
    }

    // centroids = S / count
    for (int idx = tid; idx < Kc * Ec; idx += 256) {
        const int k = idx >> 6, e = idx & 63;
        cent[((size_t)b * Kc + k) * Ec + e] = S[k][e] / msumS[k];
    }
}

// ---------------------------------------------------------------------------
// Kernel B: variance hinge. 16 blocks per batch, 1 point per thread.
// One-hot masks => only the point's own cluster contributes (mask-weighted).
// ---------------------------------------------------------------------------
__global__ void __launch_bounds__(256)
variance_kernel(const float* __restrict__ emb,
                const float* __restrict__ masks,
                const float* __restrict__ cent,
                float* __restrict__ var_acc)
{
    const int b    = blockIdx.x >> 4;
    const int nblk = blockIdx.x & 15;
    const int tid  = threadIdx.x;

    __shared__ float C[Kc][Ec + 1];
    for (int i = tid; i < Kc * Ec; i += 256)
        C[i >> 6][i & 63] = cent[(size_t)b * Kc * Ec + i];
    __syncthreads();

    const int n = nblk * 256 + tid;
    const float* __restrict__ x    = emb   + ((size_t)b * Nc + n) * Ec;
    const float* __restrict__ mrow = masks + ((size_t)b * Nc + n) * Kc;

    // locate the one-hot cluster (keep the mask value as weight).
    int   kown = 0;
    float mval = 0.f;
#pragma unroll
    for (int k = 0; k < Kc; ++k) {
        const float mk = mrow[k];
        if (mk > mval) { mval = mk; kown = k; }
    }

    float sq = 0.f;
#pragma unroll
    for (int e = 0; e < Ec; ++e) {
        const float d = x[e] - C[kown][e];
        sq = fmaf(d, d, sq);
    }
    const float dist = (sq > 0.f) ? sqrtf(sq) : 0.f;   // safe-norm forward value
    float hl = dist - DELTA_V;
    hl = (hl > 0.f) ? hl : 0.f;
    float v = mval * hl * hl;

    __shared__ float red[256];
    red[tid] = v;
    __syncthreads();
    for (int s = 128; s > 0; s >>= 1) {
        if (tid < s) red[tid] += red[tid + s];
        __syncthreads();
    }
    if (tid == 0) atomicAdd(&var_acc[b], red[0]);
}

// ---------------------------------------------------------------------------
// Kernel C: pairwise centroid hinge + centroid-norm regularizer. B blocks.
// ---------------------------------------------------------------------------
__global__ void __launch_bounds__(256)
dist_reg_kernel(const float* __restrict__ cent,
                float* __restrict__ Ld,
                float* __restrict__ Lr)
{
    const int b   = blockIdx.x;
    const int tid = threadIdx.x;

    __shared__ float C[Kc][Ec + 1];
    for (int i = tid; i < Kc * Ec; i += 256)
        C[i >> 6][i & 63] = cent[(size_t)b * Kc * Ec + i];
    __syncthreads();

    float hsum = 0.f;
    for (int p = tid; p < Kc * Kc; p += 256) {
        const int i = p >> 5, j = p & 31;
        if (i != j) {
            float sq = 0.f;
#pragma unroll
            for (int e = 0; e < Ec; ++e) {
                const float d = C[i][e] - C[j][e];
                sq = fmaf(d, d, sq);
            }
            const float cn = (sq > 0.f) ? sqrtf(sq) : 0.f;
            float hl = MARGIN_D - cn;
            hl = (hl > 0.f) ? hl : 0.f;
            hsum += hl * hl;
        }
    }

    float rsum = 0.f;
    if (tid < Kc) {
        float sq = 0.f;
#pragma unroll
        for (int e = 0; e < Ec; ++e) {
            const float c = C[tid][e];
            sq = fmaf(c, c, sq);
        }
        rsum = (sq > 0.f) ? sqrtf(sq) : 0.f;
    }

    __shared__ float red[256];
    red[tid] = hsum;
    __syncthreads();
    for (int s = 128; s > 0; s >>= 1) {
        if (tid < s) red[tid] += red[tid + s];
        __syncthreads();
    }
    if (tid == 0) Ld[b] = red[0] / (float)(Kc * (Kc - 1));
    __syncthreads();

    red[tid] = rsum;
    __syncthreads();
    for (int s = 128; s > 0; s >>= 1) {
        if (tid < s) red[tid] += red[tid + s];
        __syncthreads();
    }
    if (tid == 0) Lr[b] = red[0] / (float)Kc;
}

// ---------------------------------------------------------------------------
// Kernel D: final scalar.
// ---------------------------------------------------------------------------
__global__ void finalize_kernel(const float* __restrict__ var_acc,
                                const float* __restrict__ msum_tot,
                                const float* __restrict__ Ld,
                                const float* __restrict__ Lr,
                                float* __restrict__ out)
{
    if (threadIdx.x == 0) {
        float lv = 0.f, ld = 0.f, lr = 0.f;
#pragma unroll
        for (int b = 0; b < Bc; ++b) {
            lv += var_acc[b] / msum_tot[b];
            ld += Ld[b];
            lr += Lr[b];
        }
        out[0] = (lv + ld + GAMMA_W * lr) * (1.0f / (float)Bc);
    }
}

extern "C" void kernel_launch(void* const* d_in, const int* in_sizes, int n_in,
                              void* d_out, int out_size, void* d_ws, size_t ws_size,
                              hipStream_t stream)
{
    const float* emb   = (const float*)d_in[0];   // [8,4096,64] f32
    const float* masks = (const float*)d_in[1];   // [8,4096,32] f32
    (void)d_in; (void)in_sizes; (void)n_in; (void)ws_size; (void)out_size;

    float* ws       = (float*)d_ws;
    float* cent     = ws;                          // B*K*E = 16384 floats
    float* var_acc  = ws + (size_t)Bc * Kc * Ec;   // +8
    float* msum_tot = var_acc + Bc;                // +8
    float* Ld       = msum_tot + Bc;               // +8
    float* Lr       = Ld + Bc;                     // +8

    centroid_wmma_kernel<<<Bc, 256, 0, stream>>>(emb, masks, cent, msum_tot, var_acc);
    variance_kernel<<<Bc * (Nc / 256), 256, 0, stream>>>(emb, masks, cent, var_acc);
    dist_reg_kernel<<<Bc, 256, 0, stream>>>(cent, Ld, Lr);
    finalize_kernel<<<1, 32, 0, stream>>>(var_acc, msum_tot, Ld, Lr, (float*)d_out);
}
